// SlayerNet_2199023255633
// MI455X (gfx1250) — matrix operand order repaired and checked by reading, not compiled, not run on hardware
//
#include <hip/hip_runtime.h>
#include <hip/hip_bf16.h>

typedef __attribute__((ext_vector_type(16))) _Float16 v16h;
typedef __attribute__((ext_vector_type(8)))  _Float16 v8h;
typedef __attribute__((ext_vector_type(8)))  float    v8f;

#define DECAYF 0.90483741803595952f
#define NB 64
#define TT 100
#define E1 16384   // N * 4 * 8 * 8
#define E2 8192    // N * 8 * 4 * 4

#define WMMA_F16(A, B, C) \
  __builtin_amdgcn_wmma_f32_16x16x32_f16(false, (A), false, (B), (short)0, (C), false, false)

// ---------------------------------------------------------------------------
// Prep: build fused (conv x sumpool) weights, then emit B operands directly in
// the WMMA B-fragment register order, zero-padded to 16 columns:
//   bf[hl][kstep][lane][h],  value = W[col][32*s + 16*(lane>>4) + h]  (col = lane&15)
// hl=0: f16(w) "hi";  hl=1: f16(w - hi) "lo"  (error-compensated f16 pair).
// ---------------------------------------------------------------------------
__global__ void prep_weights_kernel(const float* __restrict__ c1,
                                    const float* __restrict__ c2,
                                    const float* __restrict__ lw,
                                    _Float16* __restrict__ bf1,   // [2][4][32][16]
                                    _Float16* __restrict__ bf2,   // [2][2][32][16]
                                    _Float16* __restrict__ bf3) { // [2][4][32][16]
  __shared__ float W8[512];   // [co(4)][K = ci*64 + dy*8 + dx]
  __shared__ float W4[512];   // [co(8)][K = ci*16 + dy*4 + dx]
  __shared__ float LIN[256];  // [o(2)][K = c*16 + h*4 + w]
  const int i = threadIdx.x;  // 512 threads
  {
    // W8 = conv1(5x5) x pool(4x4) -> 8x8 kernel, stride 4
    const int co = i >> 7, rem = i & 127;
    const int ci = rem >> 6, dy = (rem >> 3) & 7, dx = rem & 7;
    float s = 0.f;
    for (int p = 0; p < 4; ++p)
      for (int q = 0; q < 4; ++q) {
        const int ky = dy - p, kx = dx - q;
        if (ky >= 0 && ky < 5 && kx >= 0 && kx < 5)
          s += c1[((co * 2 + ci) * 5 + ky) * 5 + kx];
      }
    W8[i] = s;
  }
  {
    // W4 = conv2(3x3) x pool(2x2) -> 4x4 kernel, stride 2
    const int co = i >> 6, rem = i & 63;
    const int ci = rem >> 4, dy = (rem >> 2) & 3, dx = rem & 3;
    float s = 0.f;
    for (int p = 0; p < 2; ++p)
      for (int q = 0; q < 2; ++q) {
        const int ky = dy - p, kx = dx - q;
        if (ky >= 0 && ky < 3 && kx >= 0 && kx < 3)
          s += c2[((co * 4 + ci) * 3 + ky) * 3 + kx];
      }
    W4[i] = s;
  }
  if (i < 256) LIN[i] = lw[i];
  __syncthreads();

  // bf1: 4096 halves
  for (int idx = i; idx < 4096; idx += 512) {
    const int hl = idx >> 11, s = (idx >> 9) & 3, lane = (idx >> 4) & 31, h = idx & 15;
    const int col = lane & 15, g = lane >> 4;
    const float v = (col < 4) ? W8[col * 128 + 32 * s + 16 * g + h] : 0.f;
    const _Float16 hi = (_Float16)v;
    bf1[idx] = (hl == 0) ? hi : (_Float16)(v - (float)hi);
  }
  // bf2: 2048 halves
  for (int idx = i; idx < 2048; idx += 512) {
    const int hl = idx >> 10, s = (idx >> 9) & 1, lane = (idx >> 4) & 31, h = idx & 15;
    const int col = lane & 15, g = lane >> 4;
    const float v = (col < 8) ? W4[col * 64 + 32 * s + 16 * g + h] : 0.f;
    const _Float16 hi = (_Float16)v;
    bf2[idx] = (hl == 0) ? hi : (_Float16)(v - (float)hi);
  }
  // bf3: 4096 halves
  for (int idx = i; idx < 4096; idx += 512) {
    const int hl = idx >> 11, s = (idx >> 9) & 3, lane = (idx >> 4) & 31, h = idx & 15;
    const int col = lane & 15, g = lane >> 4;
    const float v = (col < 2) ? LIN[col * 128 + 32 * s + 16 * g + h] : 0.f;
    const _Float16 hi = (_Float16)v;
    bf3[idx] = (hl == 0) ? hi : (_Float16)(v - (float)hi);
  }
}

// ---------------------------------------------------------------------------
// Stage 1: conv1(5x5,pad2)+4x4 sumpool as implicit GEMM, K=128 (2ch x 8x8).
// One block = one (n,t) image, 4 waves x one 16x16 tile.
// A halves are K-contiguous 8-runs == contiguous padded-image rows in LDS,
// so each k-step A operand is 4 x ds_load_b64.
// ---------------------------------------------------------------------------
__global__ void conv1pool_kernel(const float* __restrict__ x,
                                 const _Float16* __restrict__ bf1,
                                 const float* __restrict__ w_pool1,
                                 float* __restrict__ h1) {
  __shared__ __align__(16) _Float16 xs[2 * 36 * 40];  // ch-stride 1440, row-stride 40
  const int tid = threadIdx.x;   // 128
  const int img = blockIdx.x;    // n*100 + t
  const int n = img / TT, t = img % TT;

  for (int i = tid; i < 2880; i += 128) xs[i] = (_Float16)0.f;
  __syncthreads();
  const float* xp = x + (size_t)img * 2048;
  for (int i = tid; i < 2048; i += 128) {
    const int ci = i >> 10, rc = i & 1023, r = rc >> 5, c = rc & 31;
    xs[ci * 1440 + (r + 2) * 40 + (c + 2)] = (_Float16)xp[i];
  }
  __syncthreads();

  const int lane = tid & 31, wave = tid >> 5;
  const int g = lane >> 4, col = lane & 15, M = lane & 15;
  const int pix = wave * 16 + M;
  const int Y = pix >> 3, X = pix & 7;
  const int pbase = 4 * Y * 40 + 4 * X;      // patch origin (halves)

  const v16h* bfh = (const v16h*)bf1;        // [s][lane]
  const v16h* bfl = (const v16h*)(bf1 + 2048);

  v8f acc = {};
#pragma unroll
  for (int s = 0; s < 4; ++s) {
    const int r1 = 4 * s + 2 * g;            // patch row index ci*8+dy
    const int r2 = r1 + 1;
    const int o1 = (r1 >> 3) * 1440 + (r1 & 7) * 40 + pbase;
    const int o2 = (r2 >> 3) * 1440 + (r2 & 7) * 40 + pbase;
    union { v16h v; unsigned long long q[4]; } a;
    a.q[0] = *(const unsigned long long*)(xs + o1);
    a.q[1] = *(const unsigned long long*)(xs + o1 + 4);
    a.q[2] = *(const unsigned long long*)(xs + o2);
    a.q[3] = *(const unsigned long long*)(xs + o2 + 4);
    const v16h bh = bfh[s * 32 + lane];
    const v16h bl = bfl[s * 32 + lane];
    acc = WMMA_F16(a.v, bh, acc);
    acc = WMMA_F16(a.v, bl, acc);
  }

  const float p1 = *w_pool1;
  if (col < 4) {
    float* out = h1 + (size_t)t * E1 + n * 256 + col * 64 + wave * 16;
#pragma unroll
    for (int j = 0; j < 8; ++j) out[j + 8 * g] = acc[j] * p1;
  }
}

// ---------------------------------------------------------------------------
// Temporal scan: u = a*u + h[t]; sp = (u - r >= 1); r = a*(r + sp)
// ---------------------------------------------------------------------------
__global__ void scan_kernel(const float* __restrict__ h,
                            _Float16* __restrict__ s, int stride) {
  const int e = blockIdx.x * blockDim.x + threadIdx.x;
  float u = 0.f, r = 0.f;
  for (int t = 0; t < TT; ++t) {
    u = DECAYF * u + h[(size_t)t * stride + e];
    const float sp = (u - r >= 1.0f) ? 1.0f : 0.0f;
    r = DECAYF * (r + sp);
    s[(size_t)t * stride + e] = (_Float16)sp;
  }
}

// ---------------------------------------------------------------------------
// Stage 2: conv2(3x3,pad1)+2x2 sumpool as implicit GEMM, K=64 (4ch x 4x4).
// One wave = one (t,n) image (16 pooled pixels x 8 Cout). A operand per
// k-step = 8 x ds_load_b32 (4-half patch rows, 4-byte aligned, row-stride 12).
// ---------------------------------------------------------------------------
__global__ void conv2pool_kernel(const _Float16* __restrict__ s1,
                                 const _Float16* __restrict__ bf2,
                                 const float* __restrict__ w_pool2,
                                 float* __restrict__ h2) {
  __shared__ __align__(16) _Float16 ps[8 * 480];  // per-wave 4ch x 10row x 12
  const int tid = threadIdx.x;  // 256
  for (int i = tid; i < 3840; i += 256) ps[i] = (_Float16)0.f;
  __syncthreads();

  const int lane = tid & 31, wave = tid >> 5;
  const int img = blockIdx.x * 8 + wave;  // t*64 + n
  const int t = img >> 6, n = img & 63;
  const _Float16* sp = s1 + (size_t)t * E1 + n * 256;
  _Float16* pw = ps + wave * 480;
  for (int i = lane; i < 256; i += 32) {
    const int ci = i >> 6, r = (i >> 3) & 7, c = i & 7;
    pw[ci * 120 + (r + 1) * 12 + (c + 1)] = sp[i];
  }
  __syncthreads();

  const int g = lane >> 4, col = lane & 15, M = lane & 15;
  const int Y = M >> 2, X = M & 3;
  const int pbase = 2 * Y * 12 + 2 * X;

  const v16h* bfh = (const v16h*)bf2;
  const v16h* bfl = (const v16h*)(bf2 + 1024);

  v8f acc = {};
#pragma unroll
  for (int s = 0; s < 2; ++s) {
    union { v16h v; unsigned int d[8]; } a;
#pragma unroll
    for (int j = 0; j < 2; ++j) {
      const int ra = 8 * s + 4 * g + 2 * j;      // patch row index ci*4+dy
      const int rb = ra + 1;
      const int oa = (ra >> 2) * 120 + (ra & 3) * 12 + pbase;
      const int ob = (rb >> 2) * 120 + (rb & 3) * 12 + pbase;
      a.d[4 * j + 0] = *(const unsigned int*)(pw + oa);
      a.d[4 * j + 1] = *(const unsigned int*)(pw + oa + 2);
      a.d[4 * j + 2] = *(const unsigned int*)(pw + ob);
      a.d[4 * j + 3] = *(const unsigned int*)(pw + ob + 2);
    }
    const v16h bh = bfh[s * 32 + lane];
    const v16h bl = bfl[s * 32 + lane];
    acc = WMMA_F16(a.v, bh, acc);
    acc = WMMA_F16(a.v, bl, acc);
  }

  const float p2 = *w_pool2;
  if (col < 8) {
    float* out = h2 + (size_t)t * E2 + n * 128 + col * 16;
#pragma unroll
    for (int j = 0; j < 8; ++j) out[j + 8 * g] = acc[j] * p2;
  }
}

// ---------------------------------------------------------------------------
// Head: out[n,t,o] = sum_k s2[t][n*128+k] * lin[o][k], K=128.
// A operand per k-step = 2 x 16B-aligned global_load_b128 from s2.
// ---------------------------------------------------------------------------
__global__ void head_kernel(const _Float16* __restrict__ s2,
                            const _Float16* __restrict__ bf3,
                            float* __restrict__ out) {
  const int tid = threadIdx.x;  // 128
  const int lane = tid & 31, wave = tid >> 5;
  const int tile = blockIdx.x * 4 + wave;
  const int g = lane >> 4, col = lane & 15, M = lane & 15;
  const int row = tile * 16 + M;          // m = n*100 + t
  const int n = row / TT, t = row % TT;
  const _Float16* ap = s2 + (size_t)t * E2 + n * 128;

  const v16h* bfh = (const v16h*)bf3;
  const v16h* bfl = (const v16h*)(bf3 + 2048);

  v8f acc = {};
#pragma unroll
  for (int s = 0; s < 4; ++s) {
    union { v16h v; v8h r[2]; } a;
    a.r[0] = *(const v8h*)(ap + 32 * s + 16 * g);
    a.r[1] = *(const v8h*)(ap + 32 * s + 16 * g + 8);
    const v16h bh = bfh[s * 32 + lane];
    const v16h bl = bfl[s * 32 + lane];
    acc = WMMA_F16(a.v, bh, acc);
    acc = WMMA_F16(a.v, bl, acc);
  }

  if (col < 2) {
#pragma unroll
    for (int j = 0; j < 8; ++j) {
      const int rr = tile * 16 + j + 8 * g;
      const int nn = rr / TT, tt = rr % TT;
      out[nn * (TT * 2) + tt * 2 + col] = acc[j];
    }
  }
}

// ---------------------------------------------------------------------------
extern "C" void kernel_launch(void* const* d_in, const int* in_sizes, int n_in,
                              void* d_out, int out_size, void* d_ws, size_t ws_size,
                              hipStream_t stream) {
  (void)in_sizes; (void)n_in; (void)out_size; (void)ws_size;
  const float* x       = (const float*)d_in[0];
  const float* conv1_w = (const float*)d_in[1];
  const float* conv2_w = (const float*)d_in[2];
  const float* lin_w   = (const float*)d_in[3];
  const float* w_pool1 = (const float*)d_in[4];
  const float* w_pool2 = (const float*)d_in[5];
  float* out = (float*)d_out;

  char* ws = (char*)d_ws;
  _Float16* bf1 = (_Float16*)(ws + 0);      // 8192 B
  _Float16* bf2 = (_Float16*)(ws + 8192);   // 4096 B
  _Float16* bf3 = (_Float16*)(ws + 12288);  // 8192 B
  float*    h1  = (float*)   (ws + 24576);                               // T*E1 f32
  _Float16* s1  = (_Float16*)(ws + 24576 + 6553600);                     // T*E1 f16
  float*    h2  = (float*)   (ws + 24576 + 6553600 + 3276800);           // T*E2 f32
  _Float16* s2  = (_Float16*)(ws + 24576 + 6553600 + 3276800 + 3276800); // T*E2 f16

  prep_weights_kernel<<<1, 512, 0, stream>>>(conv1_w, conv2_w, lin_w, bf1, bf2, bf3);
  conv1pool_kernel<<<NB * TT, 128, 0, stream>>>(x, bf1, w_pool1, h1);
  scan_kernel<<<E1 / 256, 256, 0, stream>>>(h1, s1, E1);
  conv2pool_kernel<<<(TT * NB) / 8, 256, 0, stream>>>(s1, bf2, w_pool2, h2);
  scan_kernel<<<E2 / 256, 256, 0, stream>>>(h2, s2, E2);
  head_kernel<<<(NB * TT) / 16 / 4, 128, 0, stream>>>(s2, bf3, out);
}
